// SGCN_33543694581992
// MI455X (gfx1250) — compile-verified, matrix operand order checked by reading, not run on hardware
//
#include <hip/hip_runtime.h>
#include <hip/hip_bf16.h>
#include <math.h>

#define N_NODES 100000
#define N_EDGES 1600000
#define IN_DIM  64
#define HALF_D  32

typedef __attribute__((ext_vector_type(16))) _Float16 v16h;
typedef __attribute__((ext_vector_type(8)))  float    v8f;

// ---------------- workspace layout (in floats) ----------------
#define OFF_AGGP 0            // N_NODES*64
#define OFF_AGGN 6400000      // N_NODES*64
#define OFF_AGGA 12800000     // N_NODES*32  mean_agg(zp, pos)
#define OFF_AGGB 16000000     // N_NODES*32  mean_agg(zn, neg)
#define OFF_AGGC 19200000     // N_NODES*32  mean_agg(zn, pos)
#define OFF_AGGD 22400000     // N_NODES*32  mean_agg(zp, neg)
#define OFF_CNTP 25600000     // N_NODES
#define OFF_CNTN 25700000     // N_NODES
#define OFF_Z    25800000     // N_NODES*64
#define OFF_WF16 32200000     // pre-swizzled f16 weights (in float units)
#define ZERO_COUNT 25800000   // aggregates + counts must start at 0

// pre-swizzled f16 weight region sizes (halfs)
#define W1_FRAG_HALFS (4 * 2 * 32 * 16)   // 4 k-chunks, 2 tiles -> 4096
#define W2_FRAG_HALFS (3 * 2 * 32 * 16)   // 3 k-chunks, 2 tiles -> 3072

// ---------------------------------------------------------------
__global__ __launch_bounds__(256) void zero_kernel(float* p, int n) {
    int i = blockIdx.x * blockDim.x + threadIdx.x;
    if (i < n) p[i] = 0.0f;
}

// one thread per edge: bump both degree counters
__global__ __launch_bounds__(256) void count_kernel(const int* pos_dst, const int* neg_dst,
                                                    float* cntp, float* cntn) {
    int e = blockIdx.x * blockDim.x + threadIdx.x;
    if (e >= N_EDGES) return;
    unsafeAtomicAdd(&cntp[pos_dst[e]], 1.0f);
    unsafeAtomicAdd(&cntn[neg_dst[e]], 1.0f);
}

// one thread per (edge, 4-dim group): agg[dst*64+d..d+3] += feat[src*64+d..d+3]
__global__ __launch_bounds__(256) void scatter64_kernel(const float* __restrict__ feat,
                                                        const int* __restrict__ src,
                                                        const int* __restrict__ dst,
                                                        float* agg) {
    int idx = blockIdx.x * blockDim.x + threadIdx.x;
    if (idx >= N_EDGES * 16) return;
    int e = idx >> 4;
    int d = (idx & 15) << 2;
    int s = src[e], t = dst[e];
    float4 v = *(const float4*)(feat + s * 64 + d);
    float* a = agg + t * 64 + d;
    unsafeAtomicAdd(a + 0, v.x);
    unsafeAtomicAdd(a + 1, v.y);
    unsafeAtomicAdd(a + 2, v.z);
    unsafeAtomicAdd(a + 3, v.w);
}

// one thread per (edge, 4-dim group of 32): scatters both halves of z along one edge set
// aggLow += z[src][0:32], aggHigh += z[src][32:64]
__global__ __launch_bounds__(256) void scatter32x2_kernel(const float* __restrict__ z,
                                                          const int* __restrict__ src,
                                                          const int* __restrict__ dst,
                                                          float* aggLow, float* aggHigh) {
    int idx = blockIdx.x * blockDim.x + threadIdx.x;
    if (idx >= N_EDGES * 8) return;
    int e = idx >> 3;
    int d = (idx & 7) << 2;
    int s = src[e], t = dst[e];
    float4 lo = *(const float4*)(z + s * 64 + d);
    float4 hi = *(const float4*)(z + s * 64 + HALF_D + d);
    float* al = aggLow  + t * 32 + d;
    float* ah = aggHigh + t * 32 + d;
    unsafeAtomicAdd(al + 0, lo.x);
    unsafeAtomicAdd(al + 1, lo.y);
    unsafeAtomicAdd(al + 2, lo.z);
    unsafeAtomicAdd(al + 3, lo.w);
    unsafeAtomicAdd(ah + 0, hi.x);
    unsafeAtomicAdd(ah + 1, hi.y);
    unsafeAtomicAdd(ah + 2, hi.z);
    unsafeAtomicAdd(ah + 3, hi.w);
}

// divide layer-1 aggregates by max(count,1): one thread per (node, dim<64)
__global__ __launch_bounds__(256) void normalize1_kernel(float* aggp, float* aggn,
                                                         const float* cntp, const float* cntn) {
    int idx = blockIdx.x * blockDim.x + threadIdx.x;
    if (idx >= N_NODES * 64) return;
    int node = idx >> 6;
    aggp[idx] *= (1.0f / fmaxf(cntp[node], 1.0f));
    aggn[idx] *= (1.0f / fmaxf(cntn[node], 1.0f));
}

// divide layer-2 aggregates: aggA/aggC by cntp, aggB/aggD by cntn
__global__ __launch_bounds__(256) void normalize2_kernel(float* aggA, float* aggB,
                                                         float* aggC, float* aggD,
                                                         const float* cntp, const float* cntn) {
    int idx = blockIdx.x * blockDim.x + threadIdx.x;
    if (idx >= N_NODES * 32) return;
    int node = idx >> 5;
    float ip = 1.0f / fmaxf(cntp[node], 1.0f);
    float in = 1.0f / fmaxf(cntn[node], 1.0f);
    aggA[idx] *= ip;
    aggC[idx] *= ip;
    aggB[idx] *= in;
    aggD[idx] *= in;
}

// ----------------------- WMMA helpers --------------------------
// ISA 16-bit A layout (16x32 MxK): lanes 0-15 row M=lane hold K 0..7 (e=0..7)
// and K 16..23 (e=8..15); lanes 16-31 same rows, K offset +8.
__device__ __forceinline__ int kmap(int e, int half) {
    return ((e < 8) ? e : (e + 8)) + half * 8;
}

// Load a 16x32 A fragment from row-major f32 (row stride `stride`), K window at k0.
__device__ __forceinline__ v16h load_a_frag(const float* __restrict__ base, int stride,
                                            int k0, int lane) {
    int row  = lane & 15;
    int half = lane >> 4;
    const float* p = base + row * stride + k0;
    v16h a;
#pragma unroll
    for (int e = 0; e < 16; ++e) a[e] = (_Float16)p[kmap(e, half)];
    return a;
}

// Pre-swizzled B fragment: one aligned 32-byte vector load per fragment.
// Layout: wf[((c*2 + t)*32 + lane) * 16 + e]
__device__ __forceinline__ v16h load_b_pre(const _Float16* __restrict__ wf,
                                           int c, int t, int lane) {
    return *(const v16h*)(wf + (((c * 2 + t) * 32 + lane) << 4));
}

__device__ __forceinline__ v8f wmma16(v16h a, v16h b, v8f c) {
    return __builtin_amdgcn_wmma_f32_16x16x32_f16(false, a, false, b, (short)0, c,
                                                  false, false);
}

// Convert weight w[K][32] (f32 row-major) into fragment-ordered f16:
// idx -> e(4b), lane(5b), tile(1b), chunk: element (k0+kmap(e,lane>>4), (lane&15)+t*16)
__global__ __launch_bounds__(256) void prep_w_kernel(const float* __restrict__ w,
                                                     _Float16* __restrict__ wf,
                                                     int kchunks) {
    int idx = blockIdx.x * blockDim.x + threadIdx.x;
    if (idx >= kchunks * 2 * 32 * 16) return;
    int e    = idx & 15;
    int lane = (idx >> 4) & 31;
    int t    = (idx >> 9) & 1;
    int c    = idx >> 10;
    int k    = c * 32 + kmap(e, lane >> 4);
    int col  = (lane & 15) + t * 16;
    wf[idx] = (_Float16)w[k * HALF_D + col];
}

// ---------------- layer 1: z = tanh([aggp|x]@w1b+b1b || [aggn|x]@w1u+b1u) ----------------
__global__ __launch_bounds__(256) void linear1_kernel(const float* __restrict__ aggp,
                                                      const float* __restrict__ aggn,
                                                      const float* __restrict__ x,
                                                      const _Float16* __restrict__ w1bf,
                                                      const float* __restrict__ b1b,
                                                      const _Float16* __restrict__ w1uf,
                                                      const float* __restrict__ b1u,
                                                      float* __restrict__ z) {
    int wave = (blockIdx.x * blockDim.x + threadIdx.x) >> 5;
    int lane = threadIdx.x & 31;
    if (wave >= N_NODES / 16) return;           // wave-uniform: EXEC stays all-ones
    int node_base = wave * 16;

    v8f accb0 = {}, accb1 = {}, accu0 = {}, accu1 = {};

#pragma unroll
    for (int c = 0; c < 4; ++c) {               // K = 128 = 4 chunks of 32
        const float* pb = (c < 2) ? (aggp + node_base * 64) : (x + node_base * 64);
        const float* nb = (c < 2) ? (aggn + node_base * 64) : (x + node_base * 64);
        int klocal = (c & 1) * 32;
        v16h ap = load_a_frag(pb, 64, klocal, lane);
        v16h an = load_a_frag(nb, 64, klocal, lane);
        accb0 = wmma16(ap, load_b_pre(w1bf, c, 0, lane), accb0);
        accb1 = wmma16(ap, load_b_pre(w1bf, c, 1, lane), accb1);
        accu0 = wmma16(an, load_b_pre(w1uf, c, 0, lane), accu0);
        accu1 = wmma16(an, load_b_pre(w1uf, c, 1, lane), accu1);
    }

    int n = lane & 15, half = lane >> 4;
#pragma unroll
    for (int t = 0; t < 2; ++t) {
        int col = t * 16 + n;
        float bb = b1b[col], bu = b1u[col];
        const v8f& ab = t ? accb1 : accb0;
        const v8f& au = t ? accu1 : accu0;
#pragma unroll
        for (int v = 0; v < 8; ++v) {
            int row = node_base + v + half * 8;
            z[row * 64 + col]          = tanhf(ab[v] + bb);
            z[row * 64 + HALF_D + col] = tanhf(au[v] + bu);
        }
    }
}

// ---------------- layer 2: out = tanh([aggA|aggB|zp]@w2b+b2b || [aggC|aggD|zn]@w2u+b2u) ----
__global__ __launch_bounds__(256) void linear2_kernel(const float* __restrict__ aggA,
                                                      const float* __restrict__ aggB,
                                                      const float* __restrict__ aggC,
                                                      const float* __restrict__ aggD,
                                                      const float* __restrict__ z,
                                                      const _Float16* __restrict__ w2bf,
                                                      const float* __restrict__ b2b,
                                                      const _Float16* __restrict__ w2uf,
                                                      const float* __restrict__ b2u,
                                                      float* __restrict__ out) {
    int wave = (blockIdx.x * blockDim.x + threadIdx.x) >> 5;
    int lane = threadIdx.x & 31;
    if (wave >= N_NODES / 16) return;
    int node_base = wave * 16;

    v8f accb0 = {}, accb1 = {}, accu0 = {}, accu1 = {};

    const float* pb[3] = { aggA + node_base * 32, aggB + node_base * 32,
                           z + node_base * 64 };          // zp = z cols 0..31
    const float* nb[3] = { aggC + node_base * 32, aggD + node_base * 32,
                           z + node_base * 64 + HALF_D }; // zn = z cols 32..63
    const int strides[3] = { 32, 32, 64 };

#pragma unroll
    for (int c = 0; c < 3; ++c) {               // K = 96 = 3 chunks of 32
        v16h ab = load_a_frag(pb[c], strides[c], 0, lane);
        v16h au = load_a_frag(nb[c], strides[c], 0, lane);
        accb0 = wmma16(ab, load_b_pre(w2bf, c, 0, lane), accb0);
        accb1 = wmma16(ab, load_b_pre(w2bf, c, 1, lane), accb1);
        accu0 = wmma16(au, load_b_pre(w2uf, c, 0, lane), accu0);
        accu1 = wmma16(au, load_b_pre(w2uf, c, 1, lane), accu1);
    }

    int n = lane & 15, half = lane >> 4;
#pragma unroll
    for (int t = 0; t < 2; ++t) {
        int col = t * 16 + n;
        float bb = b2b[col], bu = b2u[col];
        const v8f& ab = t ? accb1 : accb0;
        const v8f& au = t ? accu1 : accu0;
#pragma unroll
        for (int v = 0; v < 8; ++v) {
            int row = node_base + v + half * 8;
            out[row * 64 + col]          = tanhf(ab[v] + bb);
            out[row * 64 + HALF_D + col] = tanhf(au[v] + bu);
        }
    }
}

// ---------------------------------------------------------------
extern "C" void kernel_launch(void* const* d_in, const int* in_sizes, int n_in,
                              void* d_out, int out_size, void* d_ws, size_t ws_size,
                              hipStream_t stream) {
    const float* x   = (const float*)d_in[0];
    const int*   pei = (const int*)  d_in[1];   // (2, N_EDGES): [src | dst]
    const int*   nei = (const int*)  d_in[2];
    const float* w1b = (const float*)d_in[3];
    const float* b1b = (const float*)d_in[4];
    const float* w1u = (const float*)d_in[5];
    const float* b1u = (const float*)d_in[6];
    const float* w2b = (const float*)d_in[7];
    const float* b2b = (const float*)d_in[8];
    const float* w2u = (const float*)d_in[9];
    const float* b2u = (const float*)d_in[10];
    float* out = (float*)d_out;
    float* ws  = (float*)d_ws;

    float* aggp = ws + OFF_AGGP;
    float* aggn = ws + OFF_AGGN;
    float* aggA = ws + OFF_AGGA;
    float* aggB = ws + OFF_AGGB;
    float* aggC = ws + OFF_AGGC;
    float* aggD = ws + OFF_AGGD;
    float* cntp = ws + OFF_CNTP;
    float* cntn = ws + OFF_CNTN;
    float* z    = ws + OFF_Z;
    _Float16* w1bf = (_Float16*)(ws + OFF_WF16);
    _Float16* w1uf = w1bf + W1_FRAG_HALFS;
    _Float16* w2bf = w1uf + W1_FRAG_HALFS;
    _Float16* w2uf = w2bf + W2_FRAG_HALFS;

    const int* pos_src = pei;
    const int* pos_dst = pei + N_EDGES;
    const int* neg_src = nei;
    const int* neg_dst = nei + N_EDGES;

    const int B = 256;
    // 0) zero accumulators + counts; pre-swizzle weights to f16 fragment order
    zero_kernel<<<(ZERO_COUNT + B - 1) / B, B, 0, stream>>>(ws, ZERO_COUNT);
    prep_w_kernel<<<(W1_FRAG_HALFS + B - 1) / B, B, 0, stream>>>(w1b, w1bf, 4);
    prep_w_kernel<<<(W1_FRAG_HALFS + B - 1) / B, B, 0, stream>>>(w1u, w1uf, 4);
    prep_w_kernel<<<(W2_FRAG_HALFS + B - 1) / B, B, 0, stream>>>(w2b, w2bf, 3);
    prep_w_kernel<<<(W2_FRAG_HALFS + B - 1) / B, B, 0, stream>>>(w2u, w2uf, 3);
    // 1) degree counts
    count_kernel<<<(N_EDGES + B - 1) / B, B, 0, stream>>>(pos_dst, neg_dst, cntp, cntn);
    // 2) layer-1 scatters (x over pos -> aggp, x over neg -> aggn)
    {
        int blocks = (N_EDGES * 16 + B - 1) / B;
        scatter64_kernel<<<blocks, B, 0, stream>>>(x, pos_src, pos_dst, aggp);
        scatter64_kernel<<<blocks, B, 0, stream>>>(x, neg_src, neg_dst, aggn);
    }
    // 3) normalize (mean)
    normalize1_kernel<<<(N_NODES * 64 + B - 1) / B, B, 0, stream>>>(aggp, aggn, cntp, cntn);
    // 4) layer-1 linears + tanh (WMMA)
    {
        int blocks = ((N_NODES / 16) * 32 + B - 1) / B;
        linear1_kernel<<<blocks, B, 0, stream>>>(aggp, aggn, x, w1bf, b1b, w1uf, b1u, z);
    }
    // 5) layer-2 scatters: pos -> (zp->aggA, zn->aggC); neg -> (zp->aggD, zn->aggB)
    {
        int blocks = (N_EDGES * 8 + B - 1) / B;
        scatter32x2_kernel<<<blocks, B, 0, stream>>>(z, pos_src, pos_dst, aggA, aggC);
        scatter32x2_kernel<<<blocks, B, 0, stream>>>(z, neg_src, neg_dst, aggD, aggB);
    }
    // 6) normalize layer-2 aggregates
    normalize2_kernel<<<(N_NODES * 32 + B - 1) / B, B, 0, stream>>>(aggA, aggB, aggC, aggD,
                                                                    cntp, cntn);
    // 7) layer-2 linears + tanh (WMMA) -> d_out
    {
        int blocks = ((N_NODES / 16) * 32 + B - 1) / B;
        linear2_kernel<<<blocks, B, 0, stream>>>(aggA, aggB, aggC, aggD, z,
                                                 w2bf, b2b, w2uf, b2u, out);
    }
}
